// AttentionRNN_22720376996316
// MI455X (gfx1250) — matrix operand order repaired and checked by reading, not compiled, not run on hardware
//
#include <hip/hip_runtime.h>

#define B_ 32
#define T_ 256
#define VOCAB_ 128
#define EMB_ 256
#define HID_ 512
#define ATTN_ 64

typedef __attribute__((ext_vector_type(16))) __bf16 v16bf;
typedef __attribute__((ext_vector_type(8)))  float  v8f;

union Frag16 { v16bf v; uint4 q[2]; };

__device__ __forceinline__ unsigned short f2bf(float f) {
  unsigned int u = __float_as_uint(f);
  u += 0x7FFFu + ((u >> 16) & 1u);        // round-to-nearest-even
  return (unsigned short)(u >> 16);
}

// Branchless tanh on the v_exp_f32/v_rcp_f32 transcendental path.
// e = exp(-2|x|) in (0,1] (no overflow), (1+e) in (1,2] (rcp is accurate).
__device__ __forceinline__ float fast_tanh(float x) {
  const float ax = __builtin_fabsf(x);
  const float e  = __expf(-2.0f * ax);
  const float t  = (1.0f - e) * __builtin_amdgcn_rcpf(1.0f + e);
  return __builtin_copysignf(t, x);
}

__device__ __forceinline__ v8f wmma_bf16(const Frag16& a, const Frag16& b, v8f c) {
  return __builtin_amdgcn_wmma_f32_16x16x32_bf16(false, a.v, false, b.v,
                                                 (short)0, c, false, false);
}

// ---------------------------------------------------------------------------
// Weight convert + transpose: W[K][N] f32  ->  WT[N][K] bf16
// ---------------------------------------------------------------------------
__global__ void k_convT(const float* __restrict__ W, unsigned short* __restrict__ WT,
                        int K, int N) {
  long idx = (long)blockIdx.x * blockDim.x + threadIdx.x;
  if (idx >= (long)K * N) return;
  int k = (int)(idx % K);
  int n = (int)(idx / K);
  WT[idx] = f2bf(W[(long)k * N + n]);
}

// ---------------------------------------------------------------------------
// Embedding gather -> bf16   (B*T, EMB)
// ---------------------------------------------------------------------------
__global__ void k_embed(const int* __restrict__ x, const float* __restrict__ table,
                        unsigned short* __restrict__ ebf) {
  long idx = (long)blockIdx.x * blockDim.x + threadIdx.x;
  if (idx >= (long)B_ * T_ * EMB_) return;
  int  e  = (int)(idx & (EMB_ - 1));
  long bq = idx >> 8;                       // EMB_ == 256
  ebf[idx] = f2bf(table[(long)x[bq] * EMB_ + e]);
}

// ---------------------------------------------------------------------------
// Generic bf16 WMMA GEMM:  C[M,N] (f32 or bf16) = A[M,K] @ BT[N,K]^T (+bias)
// One 16x16 tile per wave, 8 waves / block. blockIdx.z = batch.
// ---------------------------------------------------------------------------
template<bool OUT_BF16, bool HAS_BIAS>
__global__ __launch_bounds__(256) void k_gemm(
    const unsigned short* __restrict__ A,  int lda, long strideA,
    const unsigned short* __restrict__ BT, int ldb, long strideB,
    void* __restrict__ C, int ldc, long strideC,
    const float* __restrict__ bias,
    int M, int N, int K)
{
  const int wave = threadIdx.x >> 5;
  const int lane = threadIdx.x & 31;
  const int hlf  = lane >> 4;
  const int r    = lane & 15;
  const int mtiles = M >> 4;
  const int ntiles = N >> 4;
  const int tile = blockIdx.x * 8 + wave;
  if (tile >= mtiles * ntiles) return;      // wave-uniform: EXEC stays all-ones
  const int mt = tile % mtiles;
  const int nt = tile / mtiles;

  const unsigned short* Ab = A  + (long)blockIdx.z * strideA + (long)(mt*16 + r) * lda;
  const unsigned short* Bb = BT + (long)blockIdx.z * strideB + (long)(nt*16 + r) * ldb;

  v8f acc;
  const float init = HAS_BIAS ? bias[nt*16 + r] : 0.0f;
#pragma unroll
  for (int v = 0; v < 8; ++v) acc[v] = init;

#pragma unroll 1
  for (int k0 = 0; k0 < K; k0 += 32) {
    Frag16 a, b;
    a.q[0] = *(const uint4*)(Ab + k0 + hlf*8);
    a.q[1] = *(const uint4*)(Ab + k0 + 16 + hlf*8);
    b.q[0] = *(const uint4*)(Bb + k0 + hlf*8);
    b.q[1] = *(const uint4*)(Bb + k0 + 16 + hlf*8);
    acc = wmma_bf16(a, b, acc);
  }

  const long cbase = (long)blockIdx.z * strideC;
#pragma unroll
  for (int v = 0; v < 8; ++v) {
    const int row = mt*16 + v + hlf*8;
    const int col = nt*16 + r;
    const long off = cbase + (long)row * ldc + col;
    if (OUT_BF16) ((unsigned short*)C)[off] = f2bf(acc[v]);
    else          ((float*)C)[off]          = acc[v];
  }
}

// ---------------------------------------------------------------------------
// Persistent 2-layer RNN. 1 block, 16 waves. Each wave owns a 2(M)x2(N) block
// of 16x16 tiles (batch 32 x 32 hidden columns): 4 accumulators, A/B fragment
// reuse, 4 independent WMMAs per K-step. Hidden state (bf16) double-buffered
// in LDS; weights stream from L2 (K-loops kept rolled: no hoist, no spill).
// LDS layout: [h0 buf0][h0 buf1][h1 buf0][h1 buf1], each B_*HID_ bf16.
// ---------------------------------------------------------------------------
__global__ __launch_bounds__(512) void k_rnn(
    const float* __restrict__ X0,
    const unsigned short* __restrict__ WT_hh0,
    const unsigned short* __restrict__ WT_ih1,
    const unsigned short* __restrict__ WT_hh1,
    const float* __restrict__ b1,
    unsigned short* __restrict__ combined,
    float* __restrict__ hfin)
{
  extern __shared__ unsigned short smem[];
  for (int i = threadIdx.x; i < 4*B_*HID_; i += blockDim.x) smem[i] = 0;
  __syncthreads();

  const int wave = threadIdx.x >> 5;        // 0..15
  const int lane = threadIdx.x & 31;
  const int hlf  = lane >> 4;
  const int r    = lane & 15;
  const int col0 = (2*wave)     * 16 + r;   // N columns this wave owns
  const int col1 = (2*wave + 1) * 16 + r;

  for (int t = 0; t < T_; ++t) {
    const int cur = t & 1, nxt = cur ^ 1;
    unsigned short* h0cur = smem + cur * (B_*HID_);
    unsigned short* h0nxt = smem + nxt * (B_*HID_);
    unsigned short* h1cur = smem + (2 + cur) * (B_*HID_);
    unsigned short* h1nxt = smem + (2 + nxt) * (B_*HID_);

    // ================= layer 0: h0' = tanh(X0[t] + h0 @ W_hh0) =============
    {
      v8f acc00, acc01, acc10, acc11;       // [mt][nti]
#pragma unroll
      for (int v = 0; v < 8; ++v) {
        const int b0r = v + hlf*8;          // mt=0 rows
        const int b1r = 16 + v + hlf*8;     // mt=1 rows
        acc00[v] = X0[((long)b0r * T_ + t) * HID_ + col0];
        acc01[v] = X0[((long)b0r * T_ + t) * HID_ + col1];
        acc10[v] = X0[((long)b1r * T_ + t) * HID_ + col0];
        acc11[v] = X0[((long)b1r * T_ + t) * HID_ + col1];
      }
      const unsigned short* Ar0 = h0cur + (r)      * HID_;
      const unsigned short* Ar1 = h0cur + (16 + r) * HID_;
      const unsigned short* Bc0 = WT_hh0 + (long)col0 * HID_;
      const unsigned short* Bc1 = WT_hh0 + (long)col1 * HID_;
#pragma unroll 1
      for (int k0 = 0; k0 < HID_; k0 += 32) {
        Frag16 a0, a1, b0, b1;
        a0.q[0] = *(const uint4*)(Ar0 + k0 + hlf*8);
        a0.q[1] = *(const uint4*)(Ar0 + k0 + 16 + hlf*8);
        a1.q[0] = *(const uint4*)(Ar1 + k0 + hlf*8);
        a1.q[1] = *(const uint4*)(Ar1 + k0 + 16 + hlf*8);
        b0.q[0] = *(const uint4*)(Bc0 + k0 + hlf*8);
        b0.q[1] = *(const uint4*)(Bc0 + k0 + 16 + hlf*8);
        b1.q[0] = *(const uint4*)(Bc1 + k0 + hlf*8);
        b1.q[1] = *(const uint4*)(Bc1 + k0 + 16 + hlf*8);
        acc00 = wmma_bf16(a0, b0, acc00);
        acc01 = wmma_bf16(a0, b1, acc01);
        acc10 = wmma_bf16(a1, b0, acc10);
        acc11 = wmma_bf16(a1, b1, acc11);
      }
#pragma unroll
      for (int v = 0; v < 8; ++v) {
        const int b0r = v + hlf*8;
        const int b1r = 16 + v + hlf*8;
        const float h00 = fast_tanh(acc00[v]);
        const float h01 = fast_tanh(acc01[v]);
        const float h10 = fast_tanh(acc10[v]);
        const float h11 = fast_tanh(acc11[v]);
        h0nxt[b0r*HID_ + col0] = f2bf(h00);
        h0nxt[b0r*HID_ + col1] = f2bf(h01);
        h0nxt[b1r*HID_ + col0] = f2bf(h10);
        h0nxt[b1r*HID_ + col1] = f2bf(h11);
        if (t == T_-1) {
          hfin[(long)b0r*HID_ + col0] = h00;
          hfin[(long)b0r*HID_ + col1] = h01;
          hfin[(long)b1r*HID_ + col0] = h10;
          hfin[(long)b1r*HID_ + col1] = h11;
        }
      }
    }
    __syncthreads();

    // ============ layer 1: h1' = tanh(h0' @ W_ih1 + h1 @ W_hh1 + b1) =======
    {
      v8f acc00, acc01, acc10, acc11;
      const float bi0 = b1[col0];
      const float bi1 = b1[col1];
#pragma unroll
      for (int v = 0; v < 8; ++v) {
        acc00[v] = bi0; acc01[v] = bi1; acc10[v] = bi0; acc11[v] = bi1;
      }
      {
        const unsigned short* Ar0 = h0nxt + (r)      * HID_;
        const unsigned short* Ar1 = h0nxt + (16 + r) * HID_;
        const unsigned short* Bc0 = WT_ih1 + (long)col0 * HID_;
        const unsigned short* Bc1 = WT_ih1 + (long)col1 * HID_;
#pragma unroll 1
        for (int k0 = 0; k0 < HID_; k0 += 32) {
          Frag16 a0, a1, b0, b1;
          a0.q[0] = *(const uint4*)(Ar0 + k0 + hlf*8);
          a0.q[1] = *(const uint4*)(Ar0 + k0 + 16 + hlf*8);
          a1.q[0] = *(const uint4*)(Ar1 + k0 + hlf*8);
          a1.q[1] = *(const uint4*)(Ar1 + k0 + 16 + hlf*8);
          b0.q[0] = *(const uint4*)(Bc0 + k0 + hlf*8);
          b0.q[1] = *(const uint4*)(Bc0 + k0 + 16 + hlf*8);
          b1.q[0] = *(const uint4*)(Bc1 + k0 + hlf*8);
          b1.q[1] = *(const uint4*)(Bc1 + k0 + 16 + hlf*8);
          acc00 = wmma_bf16(a0, b0, acc00);
          acc01 = wmma_bf16(a0, b1, acc01);
          acc10 = wmma_bf16(a1, b0, acc10);
          acc11 = wmma_bf16(a1, b1, acc11);
        }
      }
      {
        const unsigned short* Ar0 = h1cur + (r)      * HID_;
        const unsigned short* Ar1 = h1cur + (16 + r) * HID_;
        const unsigned short* Bc0 = WT_hh1 + (long)col0 * HID_;
        const unsigned short* Bc1 = WT_hh1 + (long)col1 * HID_;
#pragma unroll 1
        for (int k0 = 0; k0 < HID_; k0 += 32) {
          Frag16 a0, a1, b0, b1;
          a0.q[0] = *(const uint4*)(Ar0 + k0 + hlf*8);
          a0.q[1] = *(const uint4*)(Ar0 + k0 + 16 + hlf*8);
          a1.q[0] = *(const uint4*)(Ar1 + k0 + hlf*8);
          a1.q[1] = *(const uint4*)(Ar1 + k0 + 16 + hlf*8);
          b0.q[0] = *(const uint4*)(Bc0 + k0 + hlf*8);
          b0.q[1] = *(const uint4*)(Bc0 + k0 + 16 + hlf*8);
          b1.q[0] = *(const uint4*)(Bc1 + k0 + hlf*8);
          b1.q[1] = *(const uint4*)(Bc1 + k0 + 16 + hlf*8);
          acc00 = wmma_bf16(a0, b0, acc00);
          acc01 = wmma_bf16(a0, b1, acc01);
          acc10 = wmma_bf16(a1, b0, acc10);
          acc11 = wmma_bf16(a1, b1, acc11);
        }
      }
#pragma unroll
      for (int v = 0; v < 8; ++v) {
        const int b0r = v + hlf*8;
        const int b1r = 16 + v + hlf*8;
        const float h00 = fast_tanh(acc00[v]);
        const float h01 = fast_tanh(acc01[v]);
        const float h10 = fast_tanh(acc10[v]);
        const float h11 = fast_tanh(acc11[v]);
        h1nxt[b0r*HID_ + col0] = f2bf(h00);
        h1nxt[b0r*HID_ + col1] = f2bf(h01);
        h1nxt[b1r*HID_ + col0] = f2bf(h10);
        h1nxt[b1r*HID_ + col1] = f2bf(h11);
        combined[((long)b0r * T_ + t) * (2*HID_) + col0] = f2bf(h00);
        combined[((long)b0r * T_ + t) * (2*HID_) + col1] = f2bf(h01);
        combined[((long)b1r * T_ + t) * (2*HID_) + col0] = f2bf(h10);
        combined[((long)b1r * T_ + t) * (2*HID_) + col1] = f2bf(h11);
        if (t == T_-1) {
          hfin[(long)B_*HID_ + (long)b0r*HID_ + col0] = h00;
          hfin[(long)B_*HID_ + (long)b0r*HID_ + col1] = h01;
          hfin[(long)B_*HID_ + (long)b1r*HID_ + col0] = h10;
          hfin[(long)B_*HID_ + (long)b1r*HID_ + col1] = h11;
        }
      }
    }
    __syncthreads();
  }
}

// ---------------------------------------------------------------------------
// Fused Bahdanau scores + causal softmax. One block per (b,q); thread = key k.
// weights stored bf16, zero for k>=q and for q==0.
// ---------------------------------------------------------------------------
__global__ __launch_bounds__(256) void k_attn(
    const float* __restrict__ Eh, const float* __restrict__ Es,
    const float* __restrict__ v_att, unsigned short* __restrict__ wbf)
{
  const int bq = blockIdx.x;                 // b*T + q
  const int q  = bq & (T_ - 1);
  const int b  = bq >> 8;
  const int k  = threadIdx.x;

  __shared__ float es_sh[ATTN_];
  __shared__ float v_sh[ATTN_];
  __shared__ float red[T_];
  if (k < ATTN_) { es_sh[k] = Es[(long)bq * ATTN_ + k]; v_sh[k] = v_att[k]; }
  __syncthreads();

  float s = -3.0e38f;
  if (k < q) {
    const float* ehrow = Eh + ((long)(b * T_ + k)) * ATTN_;
    float acc = 0.f;
#pragma unroll 8
    for (int a = 0; a < ATTN_; ++a)
      acc += v_sh[a] * fast_tanh(es_sh[a] + ehrow[a]);
    s = acc;
  }
  red[k] = s; __syncthreads();
  for (int off = 128; off > 0; off >>= 1) {
    if (k < off) red[k] = fmaxf(red[k], red[k + off]);
    __syncthreads();
  }
  const float smax = red[0]; __syncthreads();
  const float e = (k < q) ? __expf(s - smax) : 0.f;
  red[k] = e; __syncthreads();
  for (int off = 128; off > 0; off >>= 1) {
    if (k < off) red[k] += red[k + off];
    __syncthreads();
  }
  const float sum = red[0];
  const float w = (q > 0 && k < q) ? e / sum : 0.f;
  wbf[(long)bq * T_ + k] = f2bf(w);
}

// ---------------------------------------------------------------------------
// HbfT[b][h][k] = Hbf[b][k][h]  (from combined cols [0,512))
// ---------------------------------------------------------------------------
__global__ void k_transpose(const unsigned short* __restrict__ combined,
                            unsigned short* __restrict__ HbfT) {
  long idx = (long)blockIdx.x * blockDim.x + threadIdx.x;
  if (idx >= (long)B_ * HID_ * T_) return;
  int  kk = (int)(idx & (T_ - 1));
  long t2 = idx >> 8;
  int  h  = (int)(t2 & (HID_ - 1));
  long b  = t2 >> 9;
  HbfT[idx] = combined[((long)(b * T_ + kk)) * (2*HID_) + h];
}

// ---------------------------------------------------------------------------
extern "C" void kernel_launch(void* const* d_in, const int* in_sizes, int n_in,
                              void* d_out, int out_size, void* d_ws, size_t ws_size,
                              hipStream_t stream)
{
  (void)in_sizes; (void)n_in; (void)out_size; (void)ws_size;
  const int*   x      = (const int*)  d_in[0];
  const float* embt   = (const float*)d_in[1];
  const float* W_ih0  = (const float*)d_in[2];
  const float* W_hh0  = (const float*)d_in[3];
  const float* b0     = (const float*)d_in[4];
  const float* W_ih1  = (const float*)d_in[5];
  const float* W_hh1  = (const float*)d_in[6];
  const float* b1     = (const float*)d_in[7];
  const float* Wh_att = (const float*)d_in[8];
  const float* Ws_att = (const float*)d_in[9];
  const float* v_att  = (const float*)d_in[10];
  const float* W_out  = (const float*)d_in[11];
  const float* b_out  = (const float*)d_in[12];
  float* out = (float*)d_out;

  char* p = (char*)d_ws;
  auto alloc = [&](size_t bytes) { char* q = p; p += (bytes + 255) & ~size_t(255); return q; };
  unsigned short* combined = (unsigned short*)alloc(2ull * B_*T_ * 2*HID_);   // 16 MB
  unsigned short* ebf      = (unsigned short*)alloc(2ull * B_*T_ * EMB_);     //  4 MB
  float*          X0       = (float*)         alloc(4ull * B_*T_ * HID_);     // 16 MB
  float*          Eh       = (float*)         alloc(4ull * B_*T_ * ATTN_);    //  2 MB
  float*          Es       = (float*)         alloc(4ull * B_*T_ * ATTN_);    //  2 MB
  unsigned short* wbf      = (unsigned short*)alloc(2ull * B_*T_ * T_);       //  4 MB
  unsigned short* HbfT     = (unsigned short*)alloc(2ull * B_ * HID_ * T_);   //  8 MB
  unsigned short* WTih0    = (unsigned short*)alloc(2ull * EMB_ * HID_);
  unsigned short* WThh0    = (unsigned short*)alloc(2ull * HID_ * HID_);
  unsigned short* WTih1    = (unsigned short*)alloc(2ull * HID_ * HID_);
  unsigned short* WThh1    = (unsigned short*)alloc(2ull * HID_ * HID_);
  unsigned short* WTh      = (unsigned short*)alloc(2ull * HID_ * ATTN_);
  unsigned short* WTs      = (unsigned short*)alloc(2ull * HID_ * ATTN_);
  unsigned short* WTo      = (unsigned short*)alloc(2ull * 2*HID_ * VOCAB_);

  auto cdiv = [](long a, long b) { return (int)((a + b - 1) / b); };

  // 1) weights -> bf16, transposed
  k_convT<<<cdiv((long)EMB_*HID_,256),256,0,stream>>>(W_ih0, WTih0, EMB_, HID_);
  k_convT<<<cdiv((long)HID_*HID_,256),256,0,stream>>>(W_hh0, WThh0, HID_, HID_);
  k_convT<<<cdiv((long)HID_*HID_,256),256,0,stream>>>(W_ih1, WTih1, HID_, HID_);
  k_convT<<<cdiv((long)HID_*HID_,256),256,0,stream>>>(W_hh1, WThh1, HID_, HID_);
  k_convT<<<cdiv((long)HID_*ATTN_,256),256,0,stream>>>(Wh_att, WTh, HID_, ATTN_);
  k_convT<<<cdiv((long)HID_*ATTN_,256),256,0,stream>>>(Ws_att, WTs, HID_, ATTN_);
  k_convT<<<cdiv((long)2*HID_*VOCAB_,256),256,0,stream>>>(W_out, WTo, 2*HID_, VOCAB_);

  // 2) embedding gather
  k_embed<<<cdiv((long)B_*T_*EMB_,256),256,0,stream>>>(x, embt, ebf);

  // 3) X0 = emb @ W_ih0 + b0   (8192 x 512 x 256)
  k_gemm<false,true><<<dim3((8192/16)*(512/16)/8,1,1),256,0,stream>>>(
      ebf, EMB_, 0, WTih0, EMB_, 0, X0, HID_, 0, b0, B_*T_, HID_, EMB_);

  // 4) sequential RNN (writes combined[:, :512] and final carries)
  k_rnn<<<1,512, 4*B_*HID_*sizeof(unsigned short), stream>>>(
      X0, WThh0, WTih1, WThh1, b1, combined, out + (long)B_*T_*VOCAB_);

  // 5) attention projections  (8192 x 64 x 512)
  k_gemm<false,false><<<dim3((8192/16)*(64/16)/8,1,1),256,0,stream>>>(
      combined, 2*HID_, 0, WTh, HID_, 0, Eh, ATTN_, 0, nullptr, B_*T_, ATTN_, HID_);
  k_gemm<false,false><<<dim3((8192/16)*(64/16)/8,1,1),256,0,stream>>>(
      combined, 2*HID_, 0, WTs, HID_, 0, Es, ATTN_, 0, nullptr, B_*T_, ATTN_, HID_);

  // 6) fused scores + causal softmax
  k_attn<<<B_*T_, T_, 0, stream>>>(Eh, Es, v_att, wbf);

  // 7) H transpose for context GEMM
  k_transpose<<<cdiv((long)B_*HID_*T_,256),256,0,stream>>>(combined, HbfT);

  // 8) context = weights @ H, batched over B; bf16 out into combined[:, 512:]
  k_gemm<true,false><<<dim3((256/16)*(512/16)/8,1,B_),256,0,stream>>>(
      wbf,  T_, (long)T_*T_,
      HbfT, T_, (long)HID_*T_,
      combined + HID_, 2*HID_, (long)T_*2*HID_,
      nullptr, T_, HID_, T_);

  // 9) logits = combined @ W_out + b_out  -> d_out (fp32)
  k_gemm<false,true><<<dim3((8192/16)*(128/16)/8,1,1),256,0,stream>>>(
      combined, 2*HID_, 0, WTo, 2*HID_, 0, out, VOCAB_, 0, b_out,
      B_*T_, VOCAB_, 2*HID_);
}
